// EnergyTransformer_12120397709648
// MI455X (gfx1250) — matrix operand order repaired and checked by reading, not compiled, not run on hardware
//
#include <hip/hip_runtime.h>

// ---------------------------------------------------------------------------
// Energy Transformer on MI455X (gfx1250): all GEMMs via v_wmma_f32_16x16x32_bf16,
// Tensor Data Mover staging (double-buffered on the Hopfield GEMM).
// ---------------------------------------------------------------------------

#ifndef ET_USE_TDM
#define ET_USE_TDM 1
#endif

typedef __attribute__((ext_vector_type(16))) __bf16 v16bf;
typedef __attribute__((ext_vector_type(8)))  float  v8f;
typedef unsigned int v4u __attribute__((ext_vector_type(4)));
typedef int          v4i __attribute__((ext_vector_type(4)));
typedef int          v8i __attribute__((ext_vector_type(8)));

constexpr int Bb = 8, Nn = 1024, Dd = 768, Hh = 12, Yy = 64, Mm = 3072;
constexpr int STEPS = 12;
constexpr float ALPHA = 0.1f;
constexpr float BETAS = 0.125f;   // 1/sqrt(Y)

#define DEVFN static __device__ __forceinline__

DEVFN v8f wmma_bf16(v16bf a, v16bf b, v8f c) {
  return __builtin_amdgcn_wmma_f32_16x16x32_bf16(false, a, false, b, (short)0, c,
                                                 false, false);
}

// ---- LDS tile staging -------------------------------------------------------
template <int ROWS, int COLS>
DEVFN void stage_rm(__bf16* lds, const __bf16* src, int stride) {
  constexpr int TOT = ROWS * COLS / 8;
  for (int idx = threadIdx.x; idx < TOT; idx += 256) {
    int r = idx / (COLS / 8);
    int c = (idx % (COLS / 8)) * 8;
    *(uint4*)(lds + r * COLS + c) = *(const uint4*)(src + (size_t)r * stride + c);
  }
}

// stage [KROWS][NCOLS] k-major source transposed into LDS [NCOLS][KROWS]
template <int KROWS, int NCOLS>
DEVFN void stage_tr(__bf16* lds, const __bf16* src, int stride) {
  constexpr int TOT = KROWS * NCOLS / 8;
  for (int idx = threadIdx.x; idx < TOT; idx += 256) {
    int k = idx / (NCOLS / 8);
    int c = (idx % (NCOLS / 8)) * 8;
    uint4 d = *(const uint4*)(src + (size_t)k * stride + c);
    __bf16 tmp[8];
    __builtin_memcpy(tmp, &d, 16);
#pragma unroll
    for (int i = 0; i < 8; ++i) lds[(c + i) * KROWS + k] = tmp[i];
  }
}

// ---- TDM: 2-D tile global->LDS via Tensor Data Mover -----------------------
#if ET_USE_TDM
DEVFN void tdm_load_tile(void* lds, const __bf16* gsrc, int tile_cols,
                         int tile_rows, int stride_elems) {
  unsigned long long ga = (unsigned long long)(size_t)gsrc;
  unsigned lo = (unsigned)(size_t)lds;     // LDS byte offset (addr[31:0])
  v4u g0;
  g0[0] = 1u;                                           // count=1, user mode
  g0[1] = lo;                                           // lds_addr
  g0[2] = (unsigned)ga;                                 // global_addr[31:0]
  g0[3] = (unsigned)((ga >> 32) & 0x01FFFFFFu) | (2u << 30);  // addr[56:32], type=2
  unsigned td0 = (unsigned)stride_elems;                // tensor_dim0 (row length)
  unsigned td1 = 0x100000u;                             // tensor_dim1 (large)
  unsigned long long s0 = (unsigned long long)stride_elems;  // dim0 stride (elems)
  v8i g1;
  g1[0] = (int)(1u << 16);                              // data_size=1 -> 2 bytes
  g1[1] = (int)((td0 & 0xFFFFu) << 16);                 // tensor_dim0[15:0]
  g1[2] = (int)((td0 >> 16) | ((td1 & 0xFFFFu) << 16)); // dim0 hi | dim1 lo
  g1[3] = (int)((td1 >> 16) | ((unsigned)tile_cols << 16));  // tile_dim0
  g1[4] = (int)(unsigned)tile_rows;                     // tile_dim1 (tile_dim2=0)
  g1[5] = (int)(unsigned)(s0 & 0xFFFFFFFFull);          // dim0_stride[31:0]
  g1[6] = (int)(unsigned)((s0 >> 32) & 0xFFFFull);      // dim0_stride[47:32]
  g1[7] = 0;
  v4i gz = {0, 0, 0, 0};
#if defined(__clang_major__) && (__clang_major__ >= 23)
  v8i gz8 = {0, 0, 0, 0, 0, 0, 0, 0};
  __builtin_amdgcn_tensor_load_to_lds(g0, g1, gz, gz, gz8, 0);
#else
  __builtin_amdgcn_tensor_load_to_lds(g0, g1, gz, gz, 0);
#endif
}
#endif

// ---- WMMA fragment loads (documented gfx1250 16-bit layouts) ---------------
DEVFN v16bf frag_a(const __bf16* lds, int ldk) {
  int lane = threadIdx.x & 31;
  int m = lane & 15;
  int kb = (lane & 16) ? 8 : 0;
  union { v16bf v; uint4 q[2]; } f;
  f.q[0] = *(const uint4*)(lds + m * ldk + kb);
  f.q[1] = *(const uint4*)(lds + m * ldk + kb + 16);
  return f.v;
}
DEVFN v16bf frag_b(const __bf16* lds, int ldk) {
  int lane = threadIdx.x & 31;
  int n = lane & 15;
  int kb = (lane & 16) ? 8 : 0;
  union { v16bf v; uint4 q[2]; } f;
  f.q[0] = *(const uint4*)(lds + n * ldk + kb);
  f.q[1] = *(const uint4*)(lds + n * ldk + kb + 16);
  return f.v;
}
// C/D f32 frag: lane col n = lane&15; reg r -> row m = r + (lane&16 ? 8 : 0)

// ---------------------------------------------------------------------------
__global__ void __launch_bounds__(256) k_cvt(const float* __restrict__ s,
                                             __bf16* __restrict__ d, int n) {
  int i = blockIdx.x * 256 + threadIdx.x;
  if (i < n) d[i] = (__bf16)s[i];
}

__global__ void __launch_bounds__(256) k_layernorm(const float* __restrict__ x,
                                                   const float* __restrict__ gamma,
                                                   const float* __restrict__ beta,
                                                   __bf16* __restrict__ g) {
  int row = blockIdx.x;
  const float* xr = x + (size_t)row * Dd;
  int t = threadIdx.x;
  float v0 = xr[t], v1 = xr[t + 256], v2 = xr[t + 512];
  float s = v0 + v1 + v2;
  float ss = v0 * v0 + v1 * v1 + v2 * v2;
#pragma unroll
  for (int off = 16; off; off >>= 1) {
    s += __shfl_down(s, off, 32);
    ss += __shfl_down(ss, off, 32);
  }
  __shared__ float red[16];
  int w = t >> 5, lane = t & 31;
  if (lane == 0) { red[w] = s; red[8 + w] = ss; }
  __syncthreads();
  if (t == 0) {
    float a = 0.f, b = 0.f;
#pragma unroll
    for (int i = 0; i < 8; ++i) { a += red[i]; b += red[8 + i]; }
    red[0] = a; red[1] = b;
  }
  __syncthreads();
  float mean = red[0] * (1.f / Dd);
  float var = red[1] * (1.f / Dd) - mean * mean;
  float rs = rsqrtf(var + 1e-5f);
  __bf16* gr = g + (size_t)row * Dd;
  gr[t]       = (__bf16)(gamma[t] * (v0 - mean) * rs + beta[t]);
  gr[t + 256] = (__bf16)(gamma[t + 256] * (v1 - mean) * rs + beta[t + 256]);
  gr[t + 512] = (__bf16)(gamma[t + 512] * (v2 - mean) * rs + beta[t + 512]);
}

// Q/K projection: out[b,h,n,y] = sum_d g[b,n,d]*w[h,y,d]  (128x64 tile, K-step 64)
__global__ void __launch_bounds__(256) k_qkproj(const __bf16* __restrict__ g,
                                                const __bf16* __restrict__ w,
                                                __bf16* __restrict__ out) {
  __shared__ __align__(16) __bf16 sA[128 * 64];
  __shared__ __align__(16) __bf16 sB[64 * 64];
  int n0 = blockIdx.x * 128;
  int bh = blockIdx.y, b = bh / Hh, h = bh % Hh;
  const __bf16* gA = g + ((size_t)b * Nn + n0) * Dd;
  const __bf16* wB = w + (size_t)h * Yy * Dd;
  int wid = threadIdx.x >> 5, lane = threadIdx.x & 31;
  v8f acc[4] = {};
  for (int d0 = 0; d0 < Dd; d0 += 64) {
#if ET_USE_TDM
    if (wid == 0) {
      tdm_load_tile(sA, gA + d0, 64, 128, Dd);
      tdm_load_tile(sB, wB + d0, 64, 64, Dd);
      __builtin_amdgcn_s_wait_tensorcnt(0);
    }
#else
    stage_rm<128, 64>(sA, gA + d0, Dd);
    stage_rm<64, 64>(sB, wB + d0, Dd);
#endif
    __syncthreads();
    v16bf a0 = frag_a(sA + wid * 16 * 64, 64);
    v16bf a1 = frag_a(sA + wid * 16 * 64 + 32, 64);
#pragma unroll
    for (int cs = 0; cs < 4; ++cs) {
      acc[cs] = wmma_bf16(a0, frag_b(sB + cs * 16 * 64, 64), acc[cs]);
      acc[cs] = wmma_bf16(a1, frag_b(sB + cs * 16 * 64 + 32, 64), acc[cs]);
    }
    __syncthreads();
  }
  int col = lane & 15, rowoff = (lane & 16) ? 8 : 0;
  __bf16* o = out + ((size_t)bh * Nn + n0 + wid * 16) * Yy;
#pragma unroll
  for (int cs = 0; cs < 4; ++cs)
#pragma unroll
    for (int r = 0; r < 8; ++r)
      o[(size_t)(r + rowoff) * Yy + cs * 16 + col] = (__bf16)acc[cs][r];
}

// Softmax stats over k (128 q / block, 32 k rows per barrier)
__global__ void __launch_bounds__(256) k_stats(const __bf16* __restrict__ Qb,
                                               const __bf16* __restrict__ Kb,
                                               float* __restrict__ mbuf,
                                               float* __restrict__ zbuf) {
  __shared__ __align__(16) __bf16 sQ[128 * 64];
  __shared__ __align__(16) __bf16 sK[32 * 64];
  int q0 = blockIdx.x * 128, bh = blockIdx.y;
  const __bf16* Q = Qb + (size_t)bh * Nn * Yy;
  const __bf16* K = Kb + (size_t)bh * Nn * Yy;
  stage_rm<128, 64>(sQ, Q + (size_t)q0 * Yy, Yy);
  __syncthreads();
  int wid = threadIdx.x >> 5, lane = threadIdx.x & 31;
  v16bf bq0 = frag_b(sQ + wid * 16 * 64, 64);
  v16bf bq1 = frag_b(sQ + wid * 16 * 64 + 32, 64);
  float mi = -1e30f, li = 0.f;
  for (int k0 = 0; k0 < Nn; k0 += 32) {
    stage_rm<32, 64>(sK, K + (size_t)k0 * Yy, Yy);
    __syncthreads();
#pragma unroll
    for (int half = 0; half < 2; ++half) {
      v16bf a0 = frag_a(sK + half * 16 * 64, 64);
      v16bf a1 = frag_a(sK + half * 16 * 64 + 32, 64);
      v8f c = {};
      c = wmma_bf16(a0, bq0, c);
      c = wmma_bf16(a1, bq1, c);
      float mn = mi;
#pragma unroll
      for (int r = 0; r < 8; ++r) mn = fmaxf(mn, c[r] * BETAS);
      float sc = __expf(mi - mn), sum = 0.f;
#pragma unroll
      for (int r = 0; r < 8; ++r) sum += __expf(c[r] * BETAS - mn);
      li = li * sc + sum;
      mi = mn;
    }
    __syncthreads();
  }
  float mo = __shfl_xor(mi, 16, 32);
  float lo = __shfl_xor(li, 16, 32);
  float mm = fmaxf(mi, mo);
  float ll = li * __expf(mi - mm) + lo * __expf(mo - mm);
  if (lane < 16) {
    int q = q0 + wid * 16 + lane;
    mbuf[(size_t)bh * Nn + q] = mm;
    zbuf[(size_t)bh * Nn + q] = ll;
  }
}

// Gq[b,q,h*64+y] = sum_k P[k,q] K[k,y],  P = exp(beta*s - m[q]) / Z[q]
__global__ void __launch_bounds__(256) k_gq(const __bf16* __restrict__ Qb,
                                            const __bf16* __restrict__ Kb,
                                            const float* __restrict__ mbuf,
                                            const float* __restrict__ zbuf,
                                            __bf16* __restrict__ Gq) {
  __shared__ __align__(16) __bf16 sQ[128 * 64];
  __shared__ __align__(16) __bf16 sKA[32 * 64];      // [k][y]
  __shared__ __align__(16) __bf16 sKB[64 * 32];      // [y][k]
  __shared__ __align__(16) __bf16 sP[8][16 * 32];    // per-wave [q][k]
  int q0 = blockIdx.x * 128, bh = blockIdx.y, b = bh / Hh, h = bh % Hh;
  const __bf16* Q = Qb + (size_t)bh * Nn * Yy;
  const __bf16* K = Kb + (size_t)bh * Nn * Yy;
  stage_rm<128, 64>(sQ, Q + (size_t)q0 * Yy, Yy);
  __syncthreads();
  int wid = threadIdx.x >> 5, lane = threadIdx.x & 31;
  v16bf bq0 = frag_b(sQ + wid * 16 * 64, 64);
  v16bf bq1 = frag_b(sQ + wid * 16 * 64 + 32, 64);
  int qcol = q0 + wid * 16 + (lane & 15);
  float mq = mbuf[(size_t)bh * Nn + qcol];
  float rz = 1.f / zbuf[(size_t)bh * Nn + qcol];
  v8f acc[4] = {};
  for (int k0 = 0; k0 < Nn; k0 += 32) {
    stage_rm<32, 64>(sKA, K + (size_t)k0 * Yy, Yy);
    stage_tr<32, 64>(sKB, K + (size_t)k0 * Yy, Yy);
    __syncthreads();
#pragma unroll
    for (int half = 0; half < 2; ++half) {
      v16bf a0 = frag_a(sKA + half * 16 * 64, 64);
      v16bf a1 = frag_a(sKA + half * 16 * 64 + 32, 64);
      v8f s = {};
      s = wmma_bf16(a0, bq0, s);
      s = wmma_bf16(a1, bq1, s);
      int kbase = half * 16 + ((lane & 16) ? 8 : 0);
#pragma unroll
      for (int r = 0; r < 8; ++r) {
        float p = __expf(s[r] * BETAS - mq) * rz;
        sP[wid][(lane & 15) * 32 + kbase + r] = (__bf16)p;
      }
    }
    __syncthreads();
    v16bf ap = frag_a(sP[wid], 32);
#pragma unroll
    for (int ys = 0; ys < 4; ++ys)
      acc[ys] = wmma_bf16(ap, frag_b(sKB + ys * 16 * 32, 32), acc[ys]);
    __syncthreads();
  }
  int rowoff = (lane & 16) ? 8 : 0;
#pragma unroll
  for (int ys = 0; ys < 4; ++ys)
#pragma unroll
    for (int r = 0; r < 8; ++r) {
      int q = q0 + wid * 16 + r + rowoff;
      int y = ys * 16 + (lane & 15);
      Gq[((size_t)b * Nn + q) * Dd + h * Yy + y] = (__bf16)acc[ys][r];
    }
}

// Gk[b,k,h*64+y] = sum_q P[k,q] Q[q,y]
__global__ void __launch_bounds__(256) k_gk(const __bf16* __restrict__ Qb,
                                            const __bf16* __restrict__ Kb,
                                            const float* __restrict__ mbuf,
                                            const float* __restrict__ zbuf,
                                            __bf16* __restrict__ Gk) {
  __shared__ __align__(16) __bf16 sK[128 * 64];      // [k][y]
  __shared__ __align__(16) __bf16 sQ1[32 * 64];      // [q][y]
  __shared__ __align__(16) __bf16 sQ2[64 * 32];      // [y][q]
  __shared__ __align__(16) __bf16 sP[8][16 * 32];    // per-wave [k][q]
  int k0 = blockIdx.x * 128, bh = blockIdx.y, b = bh / Hh, h = bh % Hh;
  const __bf16* Q = Qb + (size_t)bh * Nn * Yy;
  const __bf16* K = Kb + (size_t)bh * Nn * Yy;
  stage_rm<128, 64>(sK, K + (size_t)k0 * Yy, Yy);
  __syncthreads();
  int wid = threadIdx.x >> 5, lane = threadIdx.x & 31;
  v16bf ak0 = frag_a(sK + wid * 16 * 64, 64);
  v16bf ak1 = frag_a(sK + wid * 16 * 64 + 32, 64);
  v8f acc[4] = {};
  for (int qq = 0; qq < Nn; qq += 32) {
    stage_rm<32, 64>(sQ1, Q + (size_t)qq * Yy, Yy);
    stage_tr<32, 64>(sQ2, Q + (size_t)qq * Yy, Yy);
    __syncthreads();
#pragma unroll
    for (int half = 0; half < 2; ++half) {
      v16bf bq0 = frag_b(sQ1 + half * 16 * 64, 64);
      v16bf bq1 = frag_b(sQ1 + half * 16 * 64 + 32, 64);
      v8f s = {};
      s = wmma_bf16(ak0, bq0, s);
      s = wmma_bf16(ak1, bq1, s);
      int qg = qq + half * 16 + (lane & 15);
      float mqv = mbuf[(size_t)bh * Nn + qg];
      float rz = 1.f / zbuf[(size_t)bh * Nn + qg];
      int krow = (lane & 16) ? 8 : 0;
#pragma unroll
      for (int r = 0; r < 8; ++r) {
        float p = __expf(s[r] * BETAS - mqv) * rz;
        sP[wid][(krow + r) * 32 + half * 16 + (lane & 15)] = (__bf16)p;
      }
    }
    __syncthreads();
    v16bf ap = frag_a(sP[wid], 32);
#pragma unroll
    for (int ys = 0; ys < 4; ++ys)
      acc[ys] = wmma_bf16(ap, frag_b(sQ2 + ys * 16 * 32, 32), acc[ys]);
    __syncthreads();
  }
  int rowoff = (lane & 16) ? 8 : 0;
#pragma unroll
  for (int ys = 0; ys < 4; ++ys)
#pragma unroll
    for (int r = 0; r < 8; ++r) {
      int k = k0 + wid * 16 + r + rowoff;
      int y = ys * 16 + (lane & 15);
      Gk[((size_t)b * Nn + k) * Dd + h * Yy + y] = (__bf16)acc[ys][r];
    }
}

// grad[n,d] = sum_hy Gq[n,hy]*Wq[hy,d] + Gk[n,hy]*Wk[hy,d]  (128x128, K-step 64)
__global__ void __launch_bounds__(256) k_backproj(const __bf16* __restrict__ Gq,
                                                  const __bf16* __restrict__ Gk,
                                                  const __bf16* __restrict__ wq,
                                                  const __bf16* __restrict__ wk,
                                                  float* __restrict__ grad) {
  __shared__ __align__(16) __bf16 sA1[128 * 64], sA2[128 * 64];
  __shared__ __align__(16) __bf16 sB1[128 * 64], sB2[128 * 64];
  int n0 = blockIdx.x * 128, d0 = blockIdx.y * 128;
  int wid = threadIdx.x >> 5, lane = threadIdx.x & 31;
  v8f acc[8] = {};
  for (int k0 = 0; k0 < Dd; k0 += 64) {
#if ET_USE_TDM
    if (wid == 0) {
      tdm_load_tile(sA1, Gq + (size_t)n0 * Dd + k0, 64, 128, Dd);
      tdm_load_tile(sA2, Gk + (size_t)n0 * Dd + k0, 64, 128, Dd);
    }
#else
    stage_rm<128, 64>(sA1, Gq + (size_t)n0 * Dd + k0, Dd);
    stage_rm<128, 64>(sA2, Gk + (size_t)n0 * Dd + k0, Dd);
#endif
    stage_tr<64, 128>(sB1, wq + (size_t)k0 * Dd + d0, Dd);
    stage_tr<64, 128>(sB2, wk + (size_t)k0 * Dd + d0, Dd);
#if ET_USE_TDM
    if (wid == 0) __builtin_amdgcn_s_wait_tensorcnt(0);
#endif
    __syncthreads();
    v16bf a10 = frag_a(sA1 + wid * 16 * 64, 64);
    v16bf a11 = frag_a(sA1 + wid * 16 * 64 + 32, 64);
    v16bf a20 = frag_a(sA2 + wid * 16 * 64, 64);
    v16bf a21 = frag_a(sA2 + wid * 16 * 64 + 32, 64);
#pragma unroll
    for (int cs = 0; cs < 8; ++cs) {
      acc[cs] = wmma_bf16(a10, frag_b(sB1 + cs * 16 * 64, 64), acc[cs]);
      acc[cs] = wmma_bf16(a11, frag_b(sB1 + cs * 16 * 64 + 32, 64), acc[cs]);
      acc[cs] = wmma_bf16(a20, frag_b(sB2 + cs * 16 * 64, 64), acc[cs]);
      acc[cs] = wmma_bf16(a21, frag_b(sB2 + cs * 16 * 64 + 32, 64), acc[cs]);
    }
    __syncthreads();
  }
  int col = lane & 15, rowoff = (lane & 16) ? 8 : 0;
#pragma unroll
  for (int cs = 0; cs < 8; ++cs)
#pragma unroll
    for (int r = 0; r < 8; ++r) {
      int n = n0 + wid * 16 + r + rowoff;
      grad[(size_t)n * Dd + d0 + cs * 16 + col] = acc[cs][r];
    }
}

// Hb[n,m] = relu( sum_d g[n,d]*xi[m,d] )  (128x128, double-buffered TDM)
__global__ void __launch_bounds__(256) k_hop1(const __bf16* __restrict__ g,
                                              const __bf16* __restrict__ xi,
                                              __bf16* __restrict__ Hb) {
  __shared__ __align__(16) __bf16 sA[2][128 * 64], sB[2][128 * 64];  // 64 KB
  int n0 = blockIdx.x * 128, m0 = blockIdx.y * 128;
  const __bf16* gA = g + (size_t)n0 * Dd;
  const __bf16* gB = xi + (size_t)m0 * Dd;
  int wid = threadIdx.x >> 5, lane = threadIdx.x & 31;
  v8f acc[8] = {};
  constexpr int NIT = Dd / 64;   // 12
#if ET_USE_TDM
  if (wid == 0) {
    tdm_load_tile(sA[0], gA, 64, 128, Dd);
    tdm_load_tile(sB[0], gB, 64, 128, Dd);
  }
#endif
  for (int i = 0; i < NIT; ++i) {
    int cur = i & 1;
#if ET_USE_TDM
    if (wid == 0) {
      if (i + 1 < NIT) {
        tdm_load_tile(sA[cur ^ 1], gA + (i + 1) * 64, 64, 128, Dd);
        tdm_load_tile(sB[cur ^ 1], gB + (i + 1) * 64, 64, 128, Dd);
        __builtin_amdgcn_s_wait_tensorcnt(2);   // pair i resident, pair i+1 in flight
      } else {
        __builtin_amdgcn_s_wait_tensorcnt(0);
      }
    }
#else
    stage_rm<128, 64>(sA[cur], gA + i * 64, Dd);
    stage_rm<128, 64>(sB[cur], gB + i * 64, Dd);
#endif
    __syncthreads();
    v16bf a0 = frag_a(sA[cur] + wid * 16 * 64, 64);
    v16bf a1 = frag_a(sA[cur] + wid * 16 * 64 + 32, 64);
#pragma unroll
    for (int cs = 0; cs < 8; ++cs) {
      acc[cs] = wmma_bf16(a0, frag_b(sB[cur] + cs * 16 * 64, 64), acc[cs]);
      acc[cs] = wmma_bf16(a1, frag_b(sB[cur] + cs * 16 * 64 + 32, 64), acc[cs]);
    }
    __syncthreads();
  }
  int col = lane & 15, rowoff = (lane & 16) ? 8 : 0;
#pragma unroll
  for (int cs = 0; cs < 8; ++cs)
#pragma unroll
    for (int r = 0; r < 8; ++r) {
      int n = n0 + wid * 16 + r + rowoff;
      Hb[(size_t)n * Mm + m0 + cs * 16 + col] = (__bf16)fmaxf(acc[cs][r], 0.f);
    }
}

// grad[n,d] += sum_m Hb[n,m]*xi[m,d]  (128x128, K-step 64)
__global__ void __launch_bounds__(256) k_hop2(const __bf16* __restrict__ Hb,
                                              const __bf16* __restrict__ xi,
                                              float* __restrict__ grad) {
  __shared__ __align__(16) __bf16 sA[128 * 64], sB[128 * 64];
  int n0 = blockIdx.x * 128, d0 = blockIdx.y * 128;
  int wid = threadIdx.x >> 5, lane = threadIdx.x & 31;
  v8f acc[8] = {};
  for (int k0 = 0; k0 < Mm; k0 += 64) {
#if ET_USE_TDM
    if (wid == 0) tdm_load_tile(sA, Hb + (size_t)n0 * Mm + k0, 64, 128, Mm);
#else
    stage_rm<128, 64>(sA, Hb + (size_t)n0 * Mm + k0, Mm);
#endif
    stage_tr<64, 128>(sB, xi + (size_t)k0 * Dd + d0, Dd);
#if ET_USE_TDM
    if (wid == 0) __builtin_amdgcn_s_wait_tensorcnt(0);
#endif
    __syncthreads();
    v16bf a0 = frag_a(sA + wid * 16 * 64, 64);
    v16bf a1 = frag_a(sA + wid * 16 * 64 + 32, 64);
#pragma unroll
    for (int cs = 0; cs < 8; ++cs) {
      acc[cs] = wmma_bf16(a0, frag_b(sB + cs * 16 * 64, 64), acc[cs]);
      acc[cs] = wmma_bf16(a1, frag_b(sB + cs * 16 * 64 + 32, 64), acc[cs]);
    }
    __syncthreads();
  }
  int col = lane & 15, rowoff = (lane & 16) ? 8 : 0;
#pragma unroll
  for (int cs = 0; cs < 8; ++cs)
#pragma unroll
    for (int r = 0; r < 8; ++r) {
      int n = n0 + wid * 16 + r + rowoff;
      grad[(size_t)n * Dd + d0 + cs * 16 + col] += acc[cs][r];
    }
}

// x_out = x_src + ALPHA * grad_pos  (energy grad is negative of grad_pos)
__global__ void __launch_bounds__(256) k_update(const float* __restrict__ xs,
                                                const float* __restrict__ grad,
                                                float* __restrict__ xd, int n) {
  int i = blockIdx.x * 256 + threadIdx.x;
  if (i < n) xd[i] = xs[i] + ALPHA * grad[i];
}

// ---------------------------------------------------------------------------
extern "C" void kernel_launch(void* const* d_in, const int* in_sizes, int n_in,
                              void* d_out, int out_size, void* d_ws, size_t ws_size,
                              hipStream_t stream) {
  const float* x     = (const float*)d_in[0];
  const float* gamma = (const float*)d_in[1];
  const float* betaL = (const float*)d_in[2];
  const float* Wq    = (const float*)d_in[3];
  const float* Wk    = (const float*)d_in[4];
  const float* Xi    = (const float*)d_in[5];
  float* xout = (float*)d_out;

  const int HYD = Hh * Yy * Dd;        // 589824
  const int MD  = Mm * Dd;             // 2359296
  const int BND = Bb * Nn * Dd;        // 6291456
  const int BHN = Bb * Hh * Nn;        // 98304
  const int BHNY = Bb * Hh * Nn * Yy;  // 6291456
  const int BNM = Bb * Nn * Mm;        // 25165824

  char* ws = (char*)d_ws;
  size_t off = 0;
  auto alloc = [&](size_t bytes) -> void* {
    void* p = ws + off;
    off += (bytes + 255) & ~(size_t)255;
    return p;
  };
  __bf16* g_b  = (__bf16*)alloc((size_t)BND * 2);
  __bf16* wq_b = (__bf16*)alloc((size_t)HYD * 2);
  __bf16* wk_b = (__bf16*)alloc((size_t)HYD * 2);
  __bf16* xi_b = (__bf16*)alloc((size_t)MD * 2);
  __bf16* Qb   = (__bf16*)alloc((size_t)BHNY * 2);
  __bf16* Kb   = (__bf16*)alloc((size_t)BHNY * 2);
  float*  mbuf = (float*)alloc((size_t)BHN * 4);
  float*  zbuf = (float*)alloc((size_t)BHN * 4);
  __bf16* Gq   = (__bf16*)alloc((size_t)BND * 2);
  __bf16* Gk   = (__bf16*)alloc((size_t)BND * 2);
  __bf16* Hb   = (__bf16*)alloc((size_t)BNM * 2);
  float*  grad = (float*)alloc((size_t)BND * 4);
  // total ~147 MB of workspace

  k_cvt<<<(HYD + 255) / 256, 256, 0, stream>>>(Wq, wq_b, HYD);
  k_cvt<<<(HYD + 255) / 256, 256, 0, stream>>>(Wk, wk_b, HYD);
  k_cvt<<<(MD + 255) / 256, 256, 0, stream>>>(Xi, xi_b, MD);

  for (int step = 0; step < STEPS; ++step) {
    const float* xsrc = (step == 0) ? x : xout;
    k_layernorm<<<Bb * Nn, 256, 0, stream>>>(xsrc, gamma, betaL, g_b);
    k_qkproj<<<dim3(Nn / 128, Bb * Hh), 256, 0, stream>>>(g_b, wq_b, Qb);
    k_qkproj<<<dim3(Nn / 128, Bb * Hh), 256, 0, stream>>>(g_b, wk_b, Kb);
    k_stats<<<dim3(Nn / 128, Bb * Hh), 256, 0, stream>>>(Qb, Kb, mbuf, zbuf);
    k_gq<<<dim3(Nn / 128, Bb * Hh), 256, 0, stream>>>(Qb, Kb, mbuf, zbuf, Gq);
    k_gk<<<dim3(Nn / 128, Bb * Hh), 256, 0, stream>>>(Qb, Kb, mbuf, zbuf, Gk);
    k_backproj<<<dim3(Bb * Nn / 128, Dd / 128), 256, 0, stream>>>(Gq, Gk, wq_b, wk_b, grad);
    k_hop1<<<dim3(Bb * Nn / 128, Mm / 128), 256, 0, stream>>>(g_b, xi_b, Hb);
    k_hop2<<<dim3(Bb * Nn / 128, Dd / 128), 256, 0, stream>>>(Hb, xi_b, grad);
    k_update<<<(BND + 255) / 256, 256, 0, stream>>>(xsrc, grad, xout, BND);
  }
}
